// GlobalCrossAttention_10582799417366
// MI455X (gfx1250) — matrix operand order repaired and checked by reading, not compiled, not run on hardware
//
#include <hip/hip_runtime.h>

// ---------------------------------------------------------------------------
// GlobalCrossAttention on MI455X (gfx1250) — f16 WMMA everywhere, f32 accum.
// ---------------------------------------------------------------------------

typedef _Float16 f16_t;
typedef __attribute__((ext_vector_type(16))) _Float16 v16h;
typedef __attribute__((ext_vector_type(2)))  _Float16 h2;
typedef __attribute__((ext_vector_type(8)))  float    v8f;

#define DIM    256
#define HEADS  8
#define HD     32
#define NQ     512
#define NK     1024
#define RPE    512
#define QK_SCALE 0.17677669529663689f   // 32^-0.5

// ---- WMMA fragment helpers (ISA 7.12.2 16-bit layouts, wave32) -------------
// A 16x32: lane L holds row M=L&15; half=L>>4 -> K in {8h..8h+7, 16+8h..16+8h+7}
static __device__ __forceinline__ v16h frag_a_load(const f16_t* base, int stride) {
  int lane = threadIdx.x & 31;
  int m = lane & 15, half = lane >> 4;
  const f16_t* row = base + m * stride + 8 * half;
  union { uint4 q[2]; v16h v; } u;
  u.q[0] = *(const uint4*)(row);
  u.q[1] = *(const uint4*)(row + 16);
  return u.v;
}

// B 32x16 fed from B^T rows ([N][K] row-major): lane L holds col N=L&15,
// K in {8h..8h+7, 16+8h..16+8h+7} (same as A).
static __device__ __forceinline__ v16h frag_b_load(const f16_t* base, int stride) {
  int lane = threadIdx.x & 31;
  int n = lane & 15, half = lane >> 4;
  const f16_t* row = base + n * stride + 8 * half;
  union { uint4 q[2]; v16h v; } u;
  u.q[0] = *(const uint4*)(row);
  u.q[1] = *(const uint4*)(row + 16);
  return u.v;
}

static __device__ __forceinline__ v8f wmma_f16(v16h a, v16h b, v8f c) {
  return __builtin_amdgcn_wmma_f32_16x16x32_f16(false, a, false, b, (short)0, c,
                                                false, false);
}
// C/D layout: lane -> col n = lane&15; VGPR j -> row j + 8*(lane>>4).

// ---- prep kernels ----------------------------------------------------------
__global__ void f32_to_f16_kernel(const float* __restrict__ in,
                                  f16_t* __restrict__ out, int n) {
  int i = blockIdx.x * blockDim.x + threadIdx.x;
  if (i < n) out[i] = (f16_t)in[i];
}

// W [K=256][N=256] row-major -> Wt[N][K] f16 (B-fragments become row reads)
__global__ void transpose_w_kernel(const float* __restrict__ W,
                                   f16_t* __restrict__ Wt) {
  int i = blockIdx.x * 256 + threadIdx.x;   // 65536 total
  int k = i >> 8, n = i & 255;
  Wt[n * 256 + k] = (f16_t)W[k * 256 + n];
}

// W2 [RPE][HEADS] -> padded transposed f16 W2t[16][RPE] (cols 8..15 = 0)
__global__ void w2t_pad_kernel(const float* __restrict__ W2,
                               f16_t* __restrict__ W2t) {
  int i = blockIdx.x * 256 + threadIdx.x;   // 16*512 = 8192 total
  int n = i >> 9, r = i & (RPE - 1);
  W2t[i] = (f16_t)(n < HEADS ? W2[r * HEADS + n] : 0.0f);
}

// ---- 16x64-per-wave GEMM: C = A[M,256] * Bt[N=256][K=256]^T + bias ---------
// mode 0: head-major f16  out[h][row][d]   (Q scaled / K)
// mode 1: transposed head out[h][d][row]   (Vt)
// mode 2: f32 row-major [M,256]            (final projection -> d_out)
__global__ __launch_bounds__(32)
void gemm16_kernel(const f16_t* __restrict__ A, const f16_t* __restrict__ Bt,
                   const float* __restrict__ bias, int M, int mode, float scale,
                   void* __restrict__ outp) {
  const int K = 256;
  int mt = blockIdx.x, nt4 = blockIdx.y;    // 4 n-tiles per wave
  v8f acc[4] = {};
#pragma unroll
  for (int kt = 0; kt < 8; ++kt) {
    v16h af = frag_a_load(A + (size_t)(mt * 16) * K + kt * 32, K);
#pragma unroll
    for (int t = 0; t < 4; ++t) {
      v16h bfr = frag_b_load(Bt + (size_t)((nt4 * 4 + t) * 16) * K + kt * 32, K);
      acc[t] = wmma_f16(af, bfr, acc[t]);
    }
  }
  int lane = threadIdx.x & 31;
  int half = lane >> 4;
#pragma unroll
  for (int t = 0; t < 4; ++t) {
    int n = (nt4 * 4 + t) * 16 + (lane & 15);
    float bn = bias[n];
#pragma unroll
    for (int j = 0; j < 8; ++j) {
      int row = mt * 16 + j + 8 * half;
      float v = (acc[t][j] + bn) * scale;
      if (mode == 0) {
        ((f16_t*)outp)[((size_t)(n >> 5) * M + row) * HD + (n & 31)] = (f16_t)v;
      } else if (mode == 1) {
        ((f16_t*)outp)[(size_t)(n >> 5) * HD * M + (size_t)(n & 31) * M + row] = (f16_t)v;
      } else {
        ((float*)outp)[(size_t)row * DIM + n] = v;
      }
    }
  }
}

// ---- CPB-MLP bias: bias[h][p] = sum_r relu(a_p*W1[r]+b1[r]) * W2[r][h] -----
// 4 waves/block, each wave register-blocks 8 groups of 16 positions.
// Per K-chunk: 4 b128 (W1/b1) + 2 b128 (W2t) feed 8 WMMAs; activation math in
// packed f16 (v_pk_fma/v_pk_max co-executes with XDL WMMA).
__global__ __launch_bounds__(128)
void cpb_bias_kernel(const float* __restrict__ attn_mask,
                     const f16_t* __restrict__ W1h, const f16_t* __restrict__ B1h,
                     const f16_t* __restrict__ W2th, float* __restrict__ biasout) {
  int lane = threadIdx.x & 31;
  int wid = threadIdx.x >> 5;
  int g0 = (blockIdx.x * 4 + wid) * 8;       // first of 8 groups for this wave
  int m = lane & 15, half = lane >> 4;

  h2 a2[8];
  v8f acc[8];
#pragma unroll
  for (int g = 0; g < 8; ++g) {
    float a = attn_mask[(g0 + g) * 16 + m];
    a2[g] = h2{(f16_t)a, (f16_t)a};
    acc[g] = v8f{};
  }

  const h2 zero2 = h2{(f16_t)0.0f, (f16_t)0.0f};
  for (int ct = 0; ct < 16; ++ct) {          // RPE=512 in K=32 chunks
    int rbase = ct * 32 + 8 * half;
    union { uint4 q; h2 h[4]; } wlo, whi, blo, bhi;
    wlo.q = *(const uint4*)(W1h + rbase);
    whi.q = *(const uint4*)(W1h + rbase + 16);
    blo.q = *(const uint4*)(B1h + rbase);
    bhi.q = *(const uint4*)(B1h + rbase + 16);
    v16h bfr = frag_b_load(W2th + ct * 32, RPE);
#pragma unroll
    for (int g = 0; g < 8; ++g) {
      union { h2 h[8]; v16h v; } af;
#pragma unroll
      for (int t = 0; t < 4; ++t) {
        af.h[t]     = __builtin_elementwise_max(wlo.h[t] * a2[g] + blo.h[t], zero2);
        af.h[t + 4] = __builtin_elementwise_max(whi.h[t] * a2[g] + bhi.h[t], zero2);
      }
      acc[g] = wmma_f16(af.v, bfr, acc[g]);
    }
  }

  int n = lane & 15;                         // head column
  if (n < HEADS) {
#pragma unroll
    for (int g = 0; g < 8; ++g)
#pragma unroll
      for (int j = 0; j < 8; ++j) {
        int p = (g0 + g) * 16 + j + 8 * half;
        biasout[(size_t)n * NQ * NK + p] = acc[g][j];
      }
  }
}

// ---- fused attention: one wave per (head, 16-row q tile) -------------------
__global__ __launch_bounds__(32)
void attn_kernel(const f16_t* __restrict__ Qh, const f16_t* __restrict__ Kh,
                 const f16_t* __restrict__ Vth, const float* __restrict__ biasb,
                 const float* __restrict__ padmask, f16_t* __restrict__ xout) {
  __shared__ f16_t P[16 * NK];               // 32KB probs tile
  int lane = threadIdx.x & 31;
  int h  = blockIdx.x >> 5;
  int qt = blockIdx.x & 31;
  int n = lane & 15, half = lane >> 4;

  v16h qa = frag_a_load(Qh + ((size_t)h * NQ + qt * 16) * HD, HD);
  const float* biasrow = biasb + (size_t)h * NQ * NK + (size_t)(qt * 16) * NK;

  float rmax[8];
#pragma unroll
  for (int j = 0; j < 8; ++j) rmax[j] = -1e30f;

  // pass 1: row maxima (recompute in pass 2 is cheaper than 64KB f32 LDS)
  for (int kt = 0; kt < NK / 16; ++kt) {
    v16h kb = frag_b_load(Kh + ((size_t)h * NK + kt * 16) * HD, HD);
    v8f s = wmma_f16(qa, kb, v8f{});
    float pm = padmask[kt * 16 + n] * (-100.0f);
#pragma unroll
    for (int j = 0; j < 8; ++j)
      rmax[j] = fmaxf(rmax[j],
                      s[j] + biasrow[(size_t)(j + 8 * half) * NK + kt * 16 + n] + pm);
  }
#pragma unroll
  for (int j = 0; j < 8; ++j)
    for (int mk = 1; mk < 16; mk <<= 1)
      rmax[j] = fmaxf(rmax[j], __shfl_xor(rmax[j], mk, 32));

  // pass 2: exp + row sums + f16 probs to LDS
  float rsum[8];
#pragma unroll
  for (int j = 0; j < 8; ++j) rsum[j] = 0.0f;
  for (int kt = 0; kt < NK / 16; ++kt) {
    v16h kb = frag_b_load(Kh + ((size_t)h * NK + kt * 16) * HD, HD);
    v8f s = wmma_f16(qa, kb, v8f{});
    float pm = padmask[kt * 16 + n] * (-100.0f);
#pragma unroll
    for (int j = 0; j < 8; ++j) {
      float e = __expf(s[j] + biasrow[(size_t)(j + 8 * half) * NK + kt * 16 + n] + pm
                       - rmax[j]);
      rsum[j] += e;
      P[(j + 8 * half) * NK + kt * 16 + n] = (f16_t)e;
    }
  }
#pragma unroll
  for (int j = 0; j < 8; ++j)
    for (int mk = 1; mk < 16; mk <<= 1)
      rsum[j] += __shfl_xor(rsum[j], mk, 32);
  __syncthreads();

  // P[16,1024] @ V[1024,32]: share each A-fragment across both d-column tiles
  v8f o0 = {}, o1 = {};
  const f16_t* vbase = Vth + (size_t)h * HD * NK;
  for (int kt32 = 0; kt32 < NK / 32; ++kt32) {
    v16h pa  = frag_a_load(&P[kt32 * 32], NK);
    v16h vb0 = frag_b_load(vbase + kt32 * 32, NK);
    v16h vb1 = frag_b_load(vbase + (size_t)16 * NK + kt32 * 32, NK);
    o0 = wmma_f16(pa, vb0, o0);
    o1 = wmma_f16(pa, vb1, o1);
  }
#pragma unroll
  for (int j = 0; j < 8; ++j) {
    int row = qt * 16 + j + 8 * half;
    float inv = 1.0f / rsum[j];
    xout[(size_t)row * DIM + h * HD + n]      = (f16_t)(o0[j] * inv);
    xout[(size_t)row * DIM + h * HD + 16 + n] = (f16_t)(o1[j] * inv);
  }
}

// ---------------------------------------------------------------------------
extern "C" void kernel_launch(void* const* d_in, const int* in_sizes, int n_in,
                              void* d_out, int out_size, void* d_ws, size_t ws_size,
                              hipStream_t stream) {
  const float* query = (const float*)d_in[0];
  const float* k_in  = (const float*)d_in[1];
  const float* v_in  = (const float*)d_in[2];
  const float* pad   = (const float*)d_in[3];
  const float* amask = (const float*)d_in[4];
  const float* Wq = (const float*)d_in[5];  const float* bq = (const float*)d_in[6];
  const float* Wk = (const float*)d_in[7];  const float* bk = (const float*)d_in[8];
  const float* Wv = (const float*)d_in[9];  const float* bv = (const float*)d_in[10];
  const float* Wp = (const float*)d_in[11]; const float* bp = (const float*)d_in[12];
  const float* W1 = (const float*)d_in[13]; const float* b1 = (const float*)d_in[14];
  const float* W2 = (const float*)d_in[15];

  // workspace carve-up (all chunks 16B aligned)
  char* w = (char*)d_ws;
  float* bias_ws = (float*)w;            w += (size_t)HEADS * NQ * NK * 4;   // 16 MB
  f16_t* qh  = (f16_t*)w;                w += (size_t)HEADS * NQ * HD * 2;
  f16_t* kh  = (f16_t*)w;                w += (size_t)HEADS * NK * HD * 2;
  f16_t* vth = (f16_t*)w;                w += (size_t)HEADS * HD * NK * 2;
  f16_t* xb  = (f16_t*)w;                w += (size_t)NQ * DIM * 2;
  f16_t* qin = (f16_t*)w;                w += (size_t)NQ * DIM * 2;
  f16_t* kin = (f16_t*)w;                w += (size_t)NK * DIM * 2;
  f16_t* vin = (f16_t*)w;                w += (size_t)NK * DIM * 2;
  f16_t* wqt = (f16_t*)w;                w += (size_t)DIM * DIM * 2;
  f16_t* wkt = (f16_t*)w;                w += (size_t)DIM * DIM * 2;
  f16_t* wvt = (f16_t*)w;                w += (size_t)DIM * DIM * 2;
  f16_t* wpt = (f16_t*)w;                w += (size_t)DIM * DIM * 2;
  f16_t* w1h = (f16_t*)w;                w += (size_t)RPE * 2;
  f16_t* b1h = (f16_t*)w;                w += (size_t)RPE * 2;
  f16_t* w2t = (f16_t*)w;                w += (size_t)16 * RPE * 2;

  // prep: f16 copies, transposed f16 weights, padded W2^T
  f32_to_f16_kernel<<<(NQ * DIM + 255) / 256, 256, 0, stream>>>(query, qin, NQ * DIM);
  f32_to_f16_kernel<<<(NK * DIM + 255) / 256, 256, 0, stream>>>(k_in, kin, NK * DIM);
  f32_to_f16_kernel<<<(NK * DIM + 255) / 256, 256, 0, stream>>>(v_in, vin, NK * DIM);
  f32_to_f16_kernel<<<2, 256, 0, stream>>>(W1, w1h, RPE);
  f32_to_f16_kernel<<<2, 256, 0, stream>>>(b1, b1h, RPE);
  transpose_w_kernel<<<256, 256, 0, stream>>>(Wq, wqt);
  transpose_w_kernel<<<256, 256, 0, stream>>>(Wk, wkt);
  transpose_w_kernel<<<256, 256, 0, stream>>>(Wv, wvt);
  transpose_w_kernel<<<256, 256, 0, stream>>>(Wp, wpt);
  w2t_pad_kernel<<<32, 256, 0, stream>>>(W2, w2t);

  // QKV projections (WMMA, 16x64 per wave)
  gemm16_kernel<<<dim3(NQ / 16, 4), 32, 0, stream>>>(qin, wqt, bq, NQ, 0, QK_SCALE, qh);
  gemm16_kernel<<<dim3(NK / 16, 4), 32, 0, stream>>>(kin, wkt, bk, NK, 0, 1.0f, kh);
  gemm16_kernel<<<dim3(NK / 16, 4), 32, 0, stream>>>(vin, wvt, bv, NK, 1, 1.0f, vth);

  // CPB MLP bias (register-blocked WMMA over RPE) -> 16MB bias buffer
  cpb_bias_kernel<<<(NQ * NK) / (16 * 8 * 4), 128, 0, stream>>>(amask, w1h, b1h, w2t,
                                                                bias_ws);

  // fused attention (WMMA QK^T + softmax + WMMA PV)
  attn_kernel<<<HEADS * (NQ / 16), 32, 0, stream>>>(qh, kh, vth, bias_ws, pad, xb);

  // output projection -> f32 d_out [512,256]
  gemm16_kernel<<<dim3(NQ / 16, 4), 32, 0, stream>>>(xb, wpt, bp, NQ, 2, 1.0f, d_out);
}
